// SelfAttention_9706626089823
// MI455X (gfx1250) — matrix operand order repaired and checked by reading, not compiled, not run on hardware
//
#include <hip/hip_runtime.h>

typedef _Float16 h8   __attribute__((ext_vector_type(8)));
typedef _Float16 v16h __attribute__((ext_vector_type(16)));
typedef float    v8f  __attribute__((ext_vector_type(8)));

#define CC    128            // channels
#define NN    4096           // H*W
#define NBAT  4              // batch
#define WROWS 160            // 16 (wq) + 16 (wk) + 128 (wv)

// ---- CDNA5 async-to-LDS helpers (ASYNCcnt-tracked DMA, no VGPR round trip) --
__device__ __forceinline__ void async_ld_b128(unsigned lds_addr, const void* g) {
    asm volatile("global_load_async_to_lds_b128 %0, %1, off"
                 :: "v"(lds_addr), "v"((unsigned long long)(uintptr_t)g) : "memory");
}
__device__ __forceinline__ void async_ld_b64(unsigned lds_addr, const void* g) {
    asm volatile("global_load_async_to_lds_b64 %0, %1, off"
                 :: "v"(lds_addr), "v"((unsigned long long)(uintptr_t)g) : "memory");
}
__device__ __forceinline__ unsigned lds_off(const void* p) {
    return (unsigned)(uintptr_t)p;   // generic LDS ptr: high 32 = aperture, low 32 = LDS addr
}

// ---------------------------------------------------------------------------
// Kernel 1: x [B][C][N] f32  ->  xT [B][N][C] f16 (B-fragment friendly)
// ---------------------------------------------------------------------------
__global__ void sa_cvt_x(const float* __restrict__ x, _Float16* __restrict__ xh) {
    int idx = blockIdx.x * blockDim.x + threadIdx.x;   // over B*C*N
    int n = idx % NN;
    int t = idx / NN;
    int c = t % CC;
    int b = t / CC;
    xh[((size_t)b * NN + n) * CC + c] = (_Float16)x[idx];
}

// ---------------------------------------------------------------------------
// Kernel 2: pack wq(16x128), wk(16x128), wv(128x128) f32 -> W'[160][128] f16
// ---------------------------------------------------------------------------
__global__ void sa_cvt_w(const float* __restrict__ wq, const float* __restrict__ wk,
                         const float* __restrict__ wv, _Float16* __restrict__ wph) {
    int idx = blockIdx.x * blockDim.x + threadIdx.x;
    if (idx >= WROWS * CC) return;
    int c = idx % CC, r = idx / CC;
    float v;
    if (r < 16)       v = wq[r * CC + c];
    else if (r < 32)  v = wk[(r - 16) * CC + c];
    else              v = wv[(r - 32) * CC + c];
    wph[idx] = (_Float16)v;
}

// ---------------------------------------------------------------------------
// Kernel 3: projections via WMMA.
//   fT [B][N][32] f16 (d=0..15 data, 16..31 zero-pad)
//   gT [B][N][32] f16 (same)
//   h  [B][C][N]  f16 (row-major over n)
// ---------------------------------------------------------------------------
__global__ __launch_bounds__(256) void sa_proj(const _Float16* __restrict__ xh,
                                               const _Float16* __restrict__ wph,
                                               _Float16* __restrict__ fT,
                                               _Float16* __restrict__ gT,
                                               _Float16* __restrict__ hv) {
    const int lane = threadIdx.x & 31;
    const int wave = threadIdx.x >> 5;
    const int b    = blockIdx.x >> 5;                    // 32 blocks per batch
    const int n0   = ((blockIdx.x & 31) * 8 + wave) * 16;
    const int mrow = lane & 15;
    const int hi   = lane >> 4;

    const _Float16* xb = xh + ((size_t)b * NN + n0) * CC;

    v8f acc[10];
#pragma unroll
    for (int i = 0; i < 10; ++i)
#pragma unroll
        for (int j = 0; j < 8; ++j) acc[i][j] = 0.0f;

#pragma unroll
    for (int kc = 0; kc < CC; kc += 32) {
        v16h bx = *(const v16h*)(xb + (size_t)mrow * CC + kc + hi * 16);
#pragma unroll
        for (int wt = 0; wt < 10; ++wt) {
            const _Float16* wr = wph + (size_t)(wt * 16 + mrow) * CC + kc + hi * 16;
            union { v16h v; h8 h[2]; } ua;
            ua.h[0] = *(const h8*)(wr);
            ua.h[1] = *(const h8*)(wr + 8);
            acc[wt] = __builtin_amdgcn_wmma_f32_16x16x32_f16(
                false, ua.v, false, bx, (short)0, acc[wt], false, false);
        }
    }

    {
        h8 p;
#pragma unroll
        for (int r = 0; r < 8; ++r) p[r] = (_Float16)acc[0][r];
        *(h8*)(fT + ((size_t)b * NN + n0 + mrow) * 32 + hi * 8) = p;
#pragma unroll
        for (int r = 0; r < 8; ++r) p[r] = (_Float16)acc[1][r];
        *(h8*)(gT + ((size_t)b * NN + n0 + mrow) * 32 + hi * 8) = p;
        if (hi == 0) {   // zero-pad d = 16..31
            h8 z;
#pragma unroll
            for (int r = 0; r < 8; ++r) z[r] = (_Float16)0.0f;
            *(h8*)(fT + ((size_t)b * NN + n0 + mrow) * 32 + 16) = z;
            *(h8*)(fT + ((size_t)b * NN + n0 + mrow) * 32 + 24) = z;
            *(h8*)(gT + ((size_t)b * NN + n0 + mrow) * 32 + 16) = z;
            *(h8*)(gT + ((size_t)b * NN + n0 + mrow) * 32 + 24) = z;
        }
    }
#pragma unroll
    for (int wt = 2; wt < 10; ++wt) {
        int cbase = (wt - 2) * 16 + hi * 8;
#pragma unroll
        for (int r = 0; r < 8; ++r)
            hv[((size_t)b * CC + cbase + r) * NN + n0 + mrow] = (_Float16)acc[wt][r];
    }
}

// ---------------------------------------------------------------------------
// Kernel 4: flash attention with double-buffered async-to-LDS staging.
//   Per 32-n step the whole block shares: f-tile [32][32] (2KB), h-tile
//   [128][32] (8KB), DMA'd via global_load_async_to_lds (3 instr / wave /
//   buffer), prefetched one step ahead, retired with s_wait_asynccnt 0x3.
// ---------------------------------------------------------------------------
__global__ __launch_bounds__(256) void sa_attn(const _Float16* __restrict__ fT,
                                               const _Float16* __restrict__ gT,
                                               const _Float16* __restrict__ hv,
                                               const float* __restrict__ x,
                                               const float* __restrict__ gamma_p,
                                               float* __restrict__ out) {
    __shared__ _Float16 lf[2][32][32];       // key tile, n-major (A-frag layout)
    __shared__ _Float16 lh[2][CC][32];       // value tile, c-major (A-frag layout)
    __shared__ _Float16 pbuf[8][16][32];     // per-wave P tile [m][n]

    const int lane = threadIdx.x & 31;
    const int wave = threadIdx.x >> 5;
    const int b    = blockIdx.x >> 5;
    const int m0   = ((blockIdx.x & 31) * 8 + wave) * 16;
    const int mcol = lane & 15;
    const int hi   = lane >> 4;
    const float gamma = gamma_p[0];

    const _Float16* fb = fT + (size_t)b * NN * 32;
    const _Float16* hb = hv + (size_t)b * CC * NN;

    // q B-fragment: col m=mcol, halves j = gT[m][hi*16 + j]
    v16h qb = *(const v16h*)(gT + ((size_t)b * NN + m0 + mcol) * 32 + hi * 16);

    v8f o[8];
#pragma unroll
    for (int i = 0; i < 8; ++i)
#pragma unroll
        for (int j = 0; j < 8; ++j) o[i][j] = 0.0f;
    float mrun = -1e30f;
    float lrun = 0.0f;

    // each wave issues exactly 3 async-load instructions per buffer
    auto stage = [&](int buf, int nb0) {
        // h tile: wave w owns c rows [w*16, w*16+16): 2x b128 (32 lanes x 16B)
#pragma unroll
        for (int j = 0; j < 2; ++j) {
            int i   = j * 32 + lane;          // 0..63
            int row = i >> 2;                 // 0..15
            int ch  = i & 3;                  // 16B chunk
            const _Float16* g = hb + (size_t)(wave * 16 + row) * NN + nb0 + ch * 8;
            async_ld_b128(lds_off(&lh[buf][wave * 16 + row][ch * 8]), g);
        }
        // f tile: wave w owns n rows [w*4, w*4+4): 1x b64 (32 lanes x 8B)
        {
            int row = lane >> 3;              // 0..3
            int ch  = lane & 7;               // 8B chunk
            const _Float16* g = fb + (size_t)(nb0 + wave * 4 + row) * 32 + ch * 4;
            async_ld_b64(lds_off(&lf[buf][wave * 4 + row][ch * 4]), g);
        }
    };

    stage(0, 0);

    for (int it = 0; it < NN / 32; ++it) {
        const int nb0 = it * 32;
        const int buf = it & 1;
        if (it + 1 < NN / 32) {
            stage(buf ^ 1, nb0 + 32);                       // prefetch next
            asm volatile("s_wait_asynccnt 0x3" ::: "memory"); // retire current
        } else {
            asm volatile("s_wait_asynccnt 0x0" ::: "memory");
        }
        __syncthreads();

        // S = A(f rows n) x B(g cols m), two 16-n tiles, from LDS
        union { v16h v; h8 h[2]; } a0, a1;
        {
            const _Float16* p0 = &lf[buf][mcol][hi * 16];
            a0.h[0] = *(const h8*)(p0);
            a0.h[1] = *(const h8*)(p0 + 8);
            const _Float16* p1 = &lf[buf][16 + mcol][hi * 16];
            a1.h[0] = *(const h8*)(p1);
            a1.h[1] = *(const h8*)(p1 + 8);
        }
        v8f s0, s1;
#pragma unroll
        for (int j = 0; j < 8; ++j) { s0[j] = 0.0f; s1[j] = 0.0f; }
        s0 = __builtin_amdgcn_wmma_f32_16x16x32_f16(false, a0.v, false, qb, (short)0, s0, false, false);
        s1 = __builtin_amdgcn_wmma_f32_16x16x32_f16(false, a1.v, false, qb, (short)0, s1, false, false);

        // online softmax over n (per lane: 16 values for fixed m; pair via xor16)
        float tmax = -1e30f;
#pragma unroll
        for (int j = 0; j < 8; ++j) { tmax = fmaxf(tmax, s0[j]); tmax = fmaxf(tmax, s1[j]); }
        tmax = fmaxf(tmax, __shfl_xor(tmax, 16, 32));
        float mnew = fmaxf(mrun, tmax);
        float corr = __expf(mrun - mnew);
        float tsum = 0.0f;
        h8 e0, e1;
#pragma unroll
        for (int j = 0; j < 8; ++j) {
            float v0 = __expf(s0[j] - mnew);
            float v1 = __expf(s1[j] - mnew);
            tsum += v0 + v1;
            e0[j] = (_Float16)v0;
            e1[j] = (_Float16)v1;
        }
        tsum += __shfl_xor(tsum, 16, 32);
        lrun = lrun * corr + tsum;
        mrun = mnew;
#pragma unroll
        for (int i = 0; i < 8; ++i)
#pragma unroll
            for (int j = 0; j < 8; ++j) o[i][j] *= corr;

        // transpose P through per-wave LDS: D-layout (lane=m, vgpr=n) -> B-layout
        *(h8*)(&pbuf[wave][mcol][hi * 8])      = e0;
        *(h8*)(&pbuf[wave][mcol][16 + hi * 8]) = e1;
        v16h pb = *(const v16h*)(&pbuf[wave][mcol][hi * 16]);

        // PV: 8 c-tiles of 16 rows, h fragments from LDS
#pragma unroll
        for (int ct = 0; ct < 8; ++ct) {
            const _Float16* hp = &lh[buf][ct * 16 + mcol][hi * 16];
            union { v16h v; h8 h[2]; } ha;
            ha.h[0] = *(const h8*)(hp);
            ha.h[1] = *(const h8*)(hp + 8);
            o[ct] = __builtin_amdgcn_wmma_f32_16x16x32_f16(
                false, ha.v, false, pb, (short)0, o[ct], false, false);
        }
        __syncthreads();   // all waves done with buf before it is re-staged
    }

    // epilogue: gamma * O / l + residual
    float inv = 1.0f / lrun;
    const float* xb = x   + (size_t)b * CC * NN;
    float*       ob = out + (size_t)b * CC * NN;
#pragma unroll
    for (int ct = 0; ct < 8; ++ct) {
        int cbase = ct * 16 + hi * 8;
#pragma unroll
        for (int r = 0; r < 8; ++r) {
            size_t off = (size_t)(cbase + r) * NN + m0 + mcol;
            ob[off] = gamma * (o[ct][r] * inv) + xb[off];
        }
    }
}

// ---------------------------------------------------------------------------
extern "C" void kernel_launch(void* const* d_in, const int* in_sizes, int n_in,
                              void* d_out, int out_size, void* d_ws, size_t ws_size,
                              hipStream_t stream) {
    (void)in_sizes; (void)n_in; (void)out_size; (void)ws_size;
    const float* x     = (const float*)d_in[0];
    const float* wq    = (const float*)d_in[1];
    const float* wk    = (const float*)d_in[2];
    const float* wv    = (const float*)d_in[3];
    const float* gamma = (const float*)d_in[4];
    float* out = (float*)d_out;

    char* ws = (char*)d_ws;
    _Float16* xh  = (_Float16*)(ws);                       // 4 MiB  : [B][N][C]
    _Float16* wph = (_Float16*)(ws + 4194304);             // 40 KiB : [160][128]
    _Float16* fT  = (_Float16*)(ws + 4235264);             // 1 MiB  : [B][N][32]
    _Float16* gT  = (_Float16*)(ws + 5283840);             // 1 MiB  : [B][N][32]
    _Float16* hv  = (_Float16*)(ws + 6332416);             // 4 MiB  : [B][C][N]

    sa_cvt_x<<<(NBAT * CC * NN) / 256, 256, 0, stream>>>(x, xh);
    sa_cvt_w<<<(WROWS * CC + 255) / 256, 256, 0, stream>>>(wq, wk, wv, wph);
    sa_proj<<<NBAT * (NN / 128), 256, 0, stream>>>(xh, wph, fT, gT, hv);
    sa_attn<<<NBAT * (NN / 128), 256, 0, stream>>>(fT, gT, hv, x, gamma, out);
}
